// ScalarVectorAttentionReadout_21440476741718
// MI455X (gfx1250) — matrix operand order repaired and checked by reading, not compiled, not run on hardware
//
#include <hip/hip_runtime.h>
#include <hip/hip_bf16.h>
#include <math.h>

typedef unsigned int u32;
typedef unsigned long long u64;
typedef __attribute__((ext_vector_type(16))) __bf16 v16bf;
typedef __attribute__((ext_vector_type(8)))  float  v8f;

union FragU { v16bf v; uint4 q[2]; u32 u[8]; };
union AccU  { v8f   v; float f[8]; };

#define NN        262144
#define GG        1024
#define IN_DIM    448
#define HID       128
#define OUT_DIM   128
#define KCH       14      // 448/32
#define KCH2      4       // 128/32

__device__ __forceinline__ unsigned short f2bf(float f) {
    u32 u = __float_as_uint(f);
    u32 r = u + 0x7FFFu + ((u >> 16) & 1u);
    return (unsigned short)(r >> 16);
}
__device__ __forceinline__ u32 f2bf2(float lo, float hi) {
    return (u32)f2bf(lo) | ((u32)f2bf(hi) << 16);
}
__device__ __forceinline__ float lrelu(float x) { return x >= 0.f ? x : 0.01f * x; }

// Async copy 16B from global (saddr + 32-bit byte offset) into LDS, ASYNCcnt.
__device__ __forceinline__ void async_g2l_b128(u32 lds_byte_addr, u32 glb_byte_off,
                                               const void* saddr) {
    asm volatile("global_load_async_to_lds_b128 %0, %1, %2"
                 :: "v"(lds_byte_addr), "v"(glb_byte_off), "s"(saddr)
                 : "memory");
}
__device__ __forceinline__ void wait_asynccnt0() {
    asm volatile("s_wait_asynccnt 0x0" ::: "memory");
}

// ---------------------------------------------------------------------------
// Kernel 0: repack f32 weights into bf16 WMMA B-fragment layout.
// Layout: dst[ ((kc*8 + nt)*32 + lane)*8 + v ] = pack(W[K0][N], W[K0+1][N])
//   where h=lane/16, c=lane%16, K0 = kc*32 + (v<4 ? h*8 + 2v : 16 + h*8 + 2(v-4)), N = nt*16 + c.
// ---------------------------------------------------------------------------
__global__ __launch_bounds__(256) void prep_kernel(
    const float* __restrict__ gW1, const float* __restrict__ mW1,
    const float* __restrict__ mW2,
    u32* __restrict__ gw1b, u32* __restrict__ mw1b, u32* __restrict__ mw2b)
{
    int idx = blockIdx.x * 256 + threadIdx.x;
    const float* src; u32* dst; int j;
    if (idx < 28672)      { src = gW1; dst = gw1b; j = idx; }
    else if (idx < 57344) { src = mW1; dst = mw1b; j = idx - 28672; }
    else                  { src = mW2; dst = mw2b; j = idx - 57344; }
    int v    = j & 7;
    int lane = (j >> 3) & 31;
    int nt   = (j >> 8) & 7;
    int kc   = j >> 11;
    int h = lane >> 4, c = lane & 15;
    int K0 = kc * 32 + ((v < 4) ? (h * 8 + 2 * v) : (16 + h * 8 + 2 * (v - 4)));
    int N  = nt * 16 + c;
    dst[j] = f2bf2(src[K0 * 128 + N], src[(K0 + 1) * 128 + N]);
}

// ---------------------------------------------------------------------------
// Kernel 1: zero segment accumulators and graph_emb.
// ---------------------------------------------------------------------------
__global__ __launch_bounds__(256) void init_kernel(
    float* __restrict__ emb, float* __restrict__ segsum, u32* __restrict__ segmaxU)
{
    int idx = blockIdx.x * 256 + threadIdx.x;
    if (idx < GG * IN_DIM) emb[idx] = 0.f;
    if (idx < GG) { segsum[idx] = 0.f; segmaxU[idx] = 0u; }
}

// ---------------------------------------------------------------------------
// Kernel 2: gate MLP. Block = 256 threads (8 waves), 128 nodes/block.
// Dynamic LDS: 128x448 bf16 x-tile (114688 B) + gW1 B-fragments (114688 B).
// gW1 fragments staged via global_load_async_to_lds_b128 (overlaps with the
// f32->bf16 x-tile conversion); WMMA main loop double-buffers A and B frags.
// ---------------------------------------------------------------------------
__global__ __launch_bounds__(256) void gate_kernel(
    const float* __restrict__ scalar, const float* __restrict__ vec,
    const u32* __restrict__ gw1b,
    const float* __restrict__ gb1, const float* __restrict__ gW2,
    const float* __restrict__ gb2, float* __restrict__ gate)
{
    extern __shared__ char smem[];
    unsigned short* lx = (unsigned short*)smem;            // [128][448] bf16
    u32*            lb = (u32*)(smem + 128 * IN_DIM * 2);  // 28672 u32

    const int tid = threadIdx.x;
    const long base = (long)blockIdx.x * 128;

    // Kick off async staging of the gW1 B-fragment table (7168 x 16B).
    {
        const u32 lb_base = (u32)(uintptr_t)lb;
        for (int idx = tid; idx < 7168; idx += 256)
            async_g2l_b128(lb_base + idx * 16, idx * 16, gw1b);
    }

    // Meanwhile convert+stage the x tile (scalar: 64 f4/row, vector: 48 f4/row).
    for (int idx = tid; idx < 128 * 64; idx += 256) {
        int r = idx >> 6, q = idx & 63;
        const float4 f = ((const float4*)(scalar + (base + r) * 256))[q];
        *(uint2*)(lx + r * IN_DIM + q * 4) =
            make_uint2(f2bf2(f.x, f.y), f2bf2(f.z, f.w));
    }
    for (int idx = tid; idx < 128 * 48; idx += 256) {
        int r = idx / 48, q = idx % 48;
        const float4 f = ((const float4*)(vec + (base + r) * 192))[q];
        *(uint2*)(lx + r * IN_DIM + 256 + q * 4) =
            make_uint2(f2bf2(f.x, f.y), f2bf2(f.z, f.w));
    }
    wait_asynccnt0();
    __syncthreads();

    const int wid = tid >> 5, lane = tid & 31;
    const int h = lane >> 4, c = lane & 15;
    const int row = wid * 16 + c;

    AccU acc[8];
#pragma unroll
    for (int i = 0; i < 8; ++i)
#pragma unroll
        for (int j = 0; j < 8; ++j) acc[i].f[j] = 0.f;

    FragU aa[2];
    {   // preload A for kc=0
        const uint4* pa = (const uint4*)(lx + row * IN_DIM + h * 8);
        aa[0].q[0] = pa[0];
        aa[0].q[1] = pa[2];
    }
#pragma unroll 2
    for (int kc = 0; kc < KCH; ++kc) {
        const int cur = kc & 1;
        if (kc < KCH - 1) {  // prefetch next kchunk's A fragment early
            const uint4* pa =
                (const uint4*)(lx + row * IN_DIM + (kc + 1) * 32 + h * 8);
            aa[cur ^ 1].q[0] = pa[0];
            aa[cur ^ 1].q[1] = pa[2];
        }
        FragU bb[2];
        {
            const uint4* pb = (const uint4*)(lb + ((kc * 8 + 0) * 32 + lane) * 8);
            bb[0].q[0] = pb[0];
            bb[0].q[1] = pb[1];
        }
#pragma unroll
        for (int nt = 0; nt < 8; ++nt) {
            if (nt < 7) {  // prefetch next ntile's B fragment
                const uint4* pb =
                    (const uint4*)(lb + ((kc * 8 + nt + 1) * 32 + lane) * 8);
                bb[(nt + 1) & 1].q[0] = pb[0];
                bb[(nt + 1) & 1].q[1] = pb[1];
            }
            acc[nt].v = __builtin_amdgcn_wmma_f32_16x16x32_bf16(
                false, aa[cur].v, false, bb[nt & 1].v, (short)0,
                acc[nt].v, false, false);
        }
    }

    // Epilogue: gate[M] = sum_N lrelu(hid[M][N] + gb1[N]) * gW2[N] + gb2.
    float partial[8];
#pragma unroll
    for (int v = 0; v < 8; ++v) partial[v] = 0.f;
#pragma unroll
    for (int nt = 0; nt < 8; ++nt) {
        int N = nt * 16 + c;
        float b1 = gb1[N], w2 = gW2[N];
#pragma unroll
        for (int v = 0; v < 8; ++v)
            partial[v] += lrelu(acc[nt].f[v] + b1) * w2;
    }
#pragma unroll
    for (int v = 0; v < 8; ++v) {
        partial[v] += __shfl_xor(partial[v], 1);
        partial[v] += __shfl_xor(partial[v], 2);
        partial[v] += __shfl_xor(partial[v], 4);
        partial[v] += __shfl_xor(partial[v], 8);
    }
    if (c == 0) {
        float b2 = gb2[0];
#pragma unroll
        for (int v = 0; v < 8; ++v)
            gate[base + wid * 16 + h * 8 + v] = partial[v] + b2;
    }
}

// ---------------------------------------------------------------------------
// Kernel 3/4: segment max (ordered-uint atomicMax) and exp + segment sum.
// ---------------------------------------------------------------------------
__device__ __forceinline__ u32 ordKey(float f) {
    u32 u = __float_as_uint(f);
    return (u & 0x80000000u) ? ~u : (u | 0x80000000u);
}
__device__ __forceinline__ float ordInv(u32 k) {
    u32 u = (k & 0x80000000u) ? (k & 0x7FFFFFFFu) : ~k;
    return __uint_as_float(u);
}

__global__ __launch_bounds__(256) void segmax_kernel(
    const float* __restrict__ gate, const int* __restrict__ batch,
    u32* __restrict__ segmaxU)
{
    long n = (long)blockIdx.x * 256 + threadIdx.x;
    atomicMax(&segmaxU[batch[n]], ordKey(gate[n]));
}

__global__ __launch_bounds__(256) void segexp_kernel(
    const float* __restrict__ gate, const int* __restrict__ batch,
    const u32* __restrict__ segmaxU, float* __restrict__ a,
    float* __restrict__ segsum)
{
    long n = (long)blockIdx.x * 256 + threadIdx.x;
    int g = batch[n];
    float av = expf(gate[n] - ordInv(segmaxU[g]));
    a[n] = av;
    atomicAdd(&segsum[g], av);
}

// ---------------------------------------------------------------------------
// Kernel 5: attention pooling. 448 threads/block (thread = feature),
// 256 sorted nodes/block; register accumulate, flush at segment boundaries.
// ---------------------------------------------------------------------------
__global__ __launch_bounds__(448) void pool_kernel(
    const float* __restrict__ scalar, const float* __restrict__ vec,
    const int* __restrict__ batch, const float* __restrict__ a,
    const float* __restrict__ segsum, float* __restrict__ emb)
{
    const int k = threadIdx.x;
    const long r0 = (long)blockIdx.x * 256;
    int   curg = batch[r0];
    float inv  = 1.0f / (segsum[curg] + 1e-16f);
    float accv = 0.f;
    for (int i = 0; i < 256; ++i) {
        long n = r0 + i;
        int g = batch[n];
        if (g != curg) {
            atomicAdd(&emb[(long)curg * IN_DIM + k], accv);
            accv = 0.f;
            curg = g;
            inv = 1.0f / (segsum[g] + 1e-16f);
        }
        float w  = a[n] * inv;
        float xv = (k < 256) ? scalar[n * 256 + k] : vec[n * 192 + (k - 256)];
        accv += w * xv;
    }
    atomicAdd(&emb[(long)curg * IN_DIM + k], accv);
}

// ---------------------------------------------------------------------------
// Kernel 6: readout MLP on graph_emb (1024x448). 16 blocks x 4 waves,
// one 16-row M-tile per wave. Two bf16 WMMA GEMMs; hidden re-fragmented
// through LDS (C-layout -> A-layout).
// ---------------------------------------------------------------------------
__global__ __launch_bounds__(128) void mlp_kernel(
    const float* __restrict__ emb, const u32* __restrict__ mw1b,
    const u32* __restrict__ mw2b, const float* __restrict__ mb1,
    const float* __restrict__ mb2, float* __restrict__ out)
{
    __shared__ unsigned short lh[4 * 16 * 128];  // 16 KB hidden staging

    const int tid = threadIdx.x, wid = tid >> 5, lane = tid & 31;
    const int h = lane >> 4, c = lane & 15;
    const int m0 = (blockIdx.x * 4 + wid) * 16;
    const int row = m0 + c;

    AccU acc[8];
#pragma unroll
    for (int i = 0; i < 8; ++i)
#pragma unroll
        for (int j = 0; j < 8; ++j) acc[i].f[j] = 0.f;

    // GEMM 1: emb(16x448) x mW1(448x128)
    for (int kc = 0; kc < KCH; ++kc) {
        FragU a;
#pragma unroll
        for (int v = 0; v < 8; ++v) {
            int K0 = kc * 32 + ((v < 4) ? (h * 8 + 2 * v) : (16 + h * 8 + 2 * (v - 4)));
            a.u[v] = f2bf2(emb[row * IN_DIM + K0], emb[row * IN_DIM + K0 + 1]);
        }
#pragma unroll
        for (int nt = 0; nt < 8; ++nt) {
            FragU b;
            const uint4* pb = (const uint4*)(mw1b + (((kc * 8 + nt) * 32) + lane) * 8);
            b.q[0] = pb[0];
            b.q[1] = pb[1];
            acc[nt].v = __builtin_amdgcn_wmma_f32_16x16x32_bf16(
                false, a.v, false, b.v, (short)0, acc[nt].v, false, false);
        }
    }

    // lrelu(+mb1) -> bf16 hidden in LDS (row-major 16x128 per wave)
#pragma unroll
    for (int nt = 0; nt < 8; ++nt) {
        int N = nt * 16 + c;
        float b1 = mb1[N];
#pragma unroll
        for (int v = 0; v < 8; ++v)
            lh[(wid * 16 + v + 8 * h) * 128 + N] = f2bf(lrelu(acc[nt].f[v] + b1));
    }
    __syncthreads();

    // GEMM 2: hidden(16x128) x mW2(128x128)
    AccU acc2[8];
#pragma unroll
    for (int i = 0; i < 8; ++i)
#pragma unroll
        for (int j = 0; j < 8; ++j) acc2[i].f[j] = 0.f;

    for (int kc = 0; kc < KCH2; ++kc) {
        FragU a;
        const uint4* pa = (const uint4*)(lh + (wid * 16 + c) * 128 + kc * 32 + h * 8);
        a.q[0] = pa[0];
        a.q[1] = pa[2];
#pragma unroll
        for (int nt = 0; nt < 8; ++nt) {
            FragU b;
            const uint4* pb = (const uint4*)(mw2b + (((kc * 8 + nt) * 32) + lane) * 8);
            b.q[0] = pb[0];
            b.q[1] = pb[1];
            acc2[nt].v = __builtin_amdgcn_wmma_f32_16x16x32_bf16(
                false, a.v, false, b.v, (short)0, acc2[nt].v, false, false);
        }
    }

#pragma unroll
    for (int nt = 0; nt < 8; ++nt) {
        int N = nt * 16 + c;
        float b2 = mb2[N];
#pragma unroll
        for (int v = 0; v < 8; ++v)
            out[(m0 + v + 8 * h) * OUT_DIM + N] = acc2[nt].f[v] + b2;
    }
}

// ---------------------------------------------------------------------------
extern "C" void kernel_launch(void* const* d_in, const int* in_sizes, int n_in,
                              void* d_out, int out_size, void* d_ws, size_t ws_size,
                              hipStream_t stream)
{
    const float* scalar = (const float*)d_in[0];
    const float* vec    = (const float*)d_in[1];
    const int*   batch  = (const int*)d_in[2];
    const float* gW1    = (const float*)d_in[3];
    const float* gb1    = (const float*)d_in[4];
    const float* gW2    = (const float*)d_in[5];
    const float* gb2    = (const float*)d_in[6];
    const float* mW1    = (const float*)d_in[7];
    const float* mb1    = (const float*)d_in[8];
    const float* mW2    = (const float*)d_in[9];
    const float* mb2    = (const float*)d_in[10];
    float* out = (float*)d_out;

    // Workspace layout (~4.2 MB)
    u32*   gw1b    = (u32*)d_ws;              // 28672 u32
    u32*   mw1b    = gw1b + 28672;            // 28672 u32
    u32*   mw2b    = mw1b + 28672;            // 8192 u32
    float* gate    = (float*)(mw2b + 8192);   // 262144 f32
    float* aarr    = gate + NN;               // 262144 f32
    float* segsum  = aarr + NN;               // 1024 f32
    u32*   segmaxU = (u32*)(segsum + GG);     // 1024 u32
    float* emb     = (float*)(segmaxU + GG);  // 1024*448 f32

    prep_kernel<<<256, 256, 0, stream>>>(gW1, mW1, mW2, gw1b, mw1b, mw2b);
    init_kernel<<<(GG * IN_DIM) / 256, 256, 0, stream>>>(emb, segsum, segmaxU);

    // 229376 B dynamic LDS: 128x448 bf16 x-tile + 28672-u32 gW1 B fragments
    gate_kernel<<<NN / 128, 256, 229376, stream>>>(scalar, vec, gw1b, gb1, gW2, gb2, gate);

    segmax_kernel<<<NN / 256, 256, 0, stream>>>(gate, batch, segmaxU);
    segexp_kernel<<<NN / 256, 256, 0, stream>>>(gate, batch, segmaxU, aarr, segsum);
    pool_kernel<<<NN / 256, 448, 0, stream>>>(scalar, vec, batch, aarr, segsum, emb);
    mlp_kernel<<<GG / 64, 128, 0, stream>>>(emb, mw1b, mw2b, mb1, mb2, out);
}
